// RAEModel_17678085390614
// MI455X (gfx1250) — matrix operand order repaired and checked by reading, not compiled, not run on hardware
//
#include <hip/hip_runtime.h>

// ---------------- model constants ----------------
#define T_     2048
#define HU_    200
#define GH_    800          // 4*HU
#define EMB_   200
#define V_     50257
#define KP_    224          // K padded to 7 * 32 for wmma f16 k=32
#define NPADV_ 50304        // 786 * 64
#define NTV_   786          // vocab n-groups of 64
#define NPADX_ 832          // 13 * 64
#define NTX_   13           // 800-col n-groups of 64

typedef _Float16 v8h  __attribute__((ext_vector_type(8)));
typedef _Float16 v16h __attribute__((ext_vector_type(16)));
typedef float    v8f  __attribute__((ext_vector_type(8)));

__device__ __forceinline__ float sigm(float x) { return 1.f / (1.f + __expf(-x)); }

// A fragment: 16x32 f16, lane = hi*16 + m. halves 0..7 -> K = ks*32 + hi*8 .. +7,
// halves 8..15 -> K = ks*32 + 16 + hi*8 .. +7  (ISA 7.12.2 16-bit A table)
__device__ __forceinline__ v16h load_a_frag(const _Float16* A, int row, int ks, int hi) {
    const _Float16* p = A + (size_t)row * KP_ + ks * 32 + hi * 8;
    v8h lo = *(const v8h*)p;
    v8h hh = *(const v8h*)(p + 16);
    return __builtin_shufflevector(lo, hh, 0,1,2,3,4,5,6,7,8,9,10,11,12,13,14,15);
}
// B fragment: 32x16 f16 from BT stored [N, KP_] row-major. lane n = lane&15,
// halves 0..7 -> K = ks*32 + hi*8 .. +7, halves 8..15 -> K = ks*32 + 16 + hi*8 .. +7 (same as A)
__device__ __forceinline__ v16h load_b_frag(const _Float16* BT, int col, int ks, int hi) {
    const _Float16* p = BT + (size_t)col * KP_ + ks * 32 + hi * 8;
    v8h lo = *(const v8h*)p;
    v8h hh = *(const v8h*)(p + 16);
    return __builtin_shufflevector(lo, hh, 0,1,2,3,4,5,6,7,8,9,10,11,12,13,14,15);
}

// ---------------- init ----------------
__global__ void zero_init(float* rowSum, float* out) {
    int i = blockIdx.x * 256 + threadIdx.x;
    if (i < T_) rowSum[i] = 0.f;
    if (i == T_) out[0] = 0.f;
}

// ---------------- embedding gather -> f16 padded [T, KP] ----------------
__global__ void embed_gather(const int* sent, const float* emb, _Float16* dst) {
    int i = blockIdx.x * 256 + threadIdx.x;
    if (i >= T_ * KP_) return;
    int t = i / KP_, k = i % KP_;
    dst[i] = (k < EMB_) ? (_Float16)emb[(size_t)sent[t] * EMB_ + k] : (_Float16)0.f;
}

// ---------------- W[K0, N0] (row-stride N0) -> f16 transposed [Npad, KP] ----------------
__global__ void wT_cvt(const float* W, _Float16* dst, int K0, int N0, int Npad) {
    int i = blockIdx.x * 256 + threadIdx.x;
    if (i >= Npad * KP_) return;
    int n = i / KP_, k = i % KP_;
    dst[i] = (k < K0 && n < N0) ? (_Float16)W[(size_t)k * N0 + n] : (_Float16)0.f;
}

// ---------------- f32 [M, HU] -> f16 padded [M, KP] ----------------
__global__ void cvt_pad(const float* src, _Float16* dst, int M) {
    int i = blockIdx.x * 256 + threadIdx.x;
    if (i >= M * KP_) return;
    int m = i / KP_, k = i % KP_;
    dst[i] = (k < HU_) ? (_Float16)src[(size_t)m * HU_ + k] : (_Float16)0.f;
}

// ---------------- C[M,N] = A[M,KP]f16 @ BT[N,KP]f16 + bias, f32 out -------------
// block = 8 waves = 8 consecutive m-tiles sharing one 64-wide n-group (B reuse in L0)
__global__ void __launch_bounds__(256) gemm_bias_f16(const _Float16* A, const _Float16* BT,
                                                     const float* bias, float* C,
                                                     int N, int NT) {
    int wave = threadIdx.x >> 5, lane = threadIdx.x & 31;
    int mt = (blockIdx.x / NT) * 8 + wave;
    int n0 = (blockIdx.x % NT) * 64;
    int m0 = mt * 16;
    int hi = lane >> 4, ln = lane & 15;
    v8f acc0 = {0,0,0,0,0,0,0,0}, acc1 = acc0, acc2 = acc0, acc3 = acc0;
    for (int ks = 0; ks < 7; ++ks) {
        v16h a  = load_a_frag(A, m0 + ln, ks, hi);
        v16h b0 = load_b_frag(BT, n0 +  0 + ln, ks, hi);
        v16h b1 = load_b_frag(BT, n0 + 16 + ln, ks, hi);
        v16h b2 = load_b_frag(BT, n0 + 32 + ln, ks, hi);
        v16h b3 = load_b_frag(BT, n0 + 48 + ln, ks, hi);
        acc0 = __builtin_amdgcn_wmma_f32_16x16x32_f16(false, a, false, b0, (short)0, acc0, false, false);
        acc1 = __builtin_amdgcn_wmma_f32_16x16x32_f16(false, a, false, b1, (short)0, acc1, false, false);
        acc2 = __builtin_amdgcn_wmma_f32_16x16x32_f16(false, a, false, b2, (short)0, acc2, false, false);
        acc3 = __builtin_amdgcn_wmma_f32_16x16x32_f16(false, a, false, b3, (short)0, acc3, false, false);
    }
    v8f* accs[4] = {&acc0, &acc1, &acc2, &acc3};
    #pragma unroll
    for (int nn = 0; nn < 4; ++nn) {
        int ng = n0 + nn * 16 + ln;
        if (ng < N) {
            float bb = bias[ng];
            #pragma unroll
            for (int r = 0; r < 8; ++r)
                C[(size_t)(m0 + r + 8 * hi) * N + ng] = (*accs[nn])[r] + bb;
        }
    }
}

// ---------------- logits GEMM fused with exp-row-sum --------------------------
__global__ void __launch_bounds__(256) logits_lse(const _Float16* A, const _Float16* BT,
                                                  const float* sb, float* rowSum) {
    int wave = threadIdx.x >> 5, lane = threadIdx.x & 31;
    int mt = (blockIdx.x / NTV_) * 8 + wave;
    int n0 = (blockIdx.x % NTV_) * 64;
    int m0 = mt * 16;
    int hi = lane >> 4, ln = lane & 15;
    v8f acc0 = {0,0,0,0,0,0,0,0}, acc1 = acc0, acc2 = acc0, acc3 = acc0;
    for (int ks = 0; ks < 7; ++ks) {
        v16h a  = load_a_frag(A, m0 + ln, ks, hi);
        v16h b0 = load_b_frag(BT, n0 +  0 + ln, ks, hi);
        v16h b1 = load_b_frag(BT, n0 + 16 + ln, ks, hi);
        v16h b2 = load_b_frag(BT, n0 + 32 + ln, ks, hi);
        v16h b3 = load_b_frag(BT, n0 + 48 + ln, ks, hi);
        if (ks < 6)  // stream next k-slab of the 22.5 MB weight panel through L2
            __builtin_prefetch(BT + (size_t)(n0 + ln) * KP_ + (ks + 1) * 32, 0, 1);
        acc0 = __builtin_amdgcn_wmma_f32_16x16x32_f16(false, a, false, b0, (short)0, acc0, false, false);
        acc1 = __builtin_amdgcn_wmma_f32_16x16x32_f16(false, a, false, b1, (short)0, acc1, false, false);
        acc2 = __builtin_amdgcn_wmma_f32_16x16x32_f16(false, a, false, b2, (short)0, acc2, false, false);
        acc3 = __builtin_amdgcn_wmma_f32_16x16x32_f16(false, a, false, b3, (short)0, acc3, false, false);
    }
    v8f* accs[4] = {&acc0, &acc1, &acc2, &acc3};
    float rs[8] = {0,0,0,0,0,0,0,0};
    #pragma unroll
    for (int nn = 0; nn < 4; ++nn) {
        int ng = n0 + nn * 16 + ln;
        bool valid = ng < V_;
        float bb = valid ? sb[ng] : 0.f;
        #pragma unroll
        for (int r = 0; r < 8; ++r) {
            float lg = (*accs[nn])[r] + bb;
            rs[r] += valid ? __expf(lg) : 0.f;   // logits ~O(1): no max-shift needed
        }
    }
    #pragma unroll
    for (int r = 0; r < 8; ++r) {
        float v = rs[r];
        v += __shfl_xor(v, 1, 32); v += __shfl_xor(v, 2, 32);
        v += __shfl_xor(v, 4, 32); v += __shfl_xor(v, 8, 32);
        if (ln == 0) atomicAdd(&rowSum[m0 + r + 8 * hi], v);
    }
}

// ---------------- sequential LSTM scan, recurrent weights resident in LDS -----
// 800 threads, dyn LDS = 200*800 f16 (Wh) + 200 f32 (h) + 800 f32 (gates) = 324 KB
__global__ void __launch_bounds__(800) lstm_layer(const float* Wfull, const float* xW,
                                                  int xw_stride, float* h_out) {
    extern __shared__ char smem[];
    _Float16* Wh  = (_Float16*)smem;                       // [200][800]
    float* hbuf   = (float*)(smem + HU_ * GH_ * 2);        // [200]
    float* gbuf   = hbuf + HU_;                            // [800]
    int tid = threadIdx.x;
    const float* Whg = Wfull + (size_t)HU_ * GH_;          // rows 200..399 = h-part
    for (int k = 0; k < HU_; ++k) Wh[k * GH_ + tid] = (_Float16)Whg[(size_t)k * GH_ + tid];
    if (tid < HU_) hbuf[tid] = 0.f;
    float c = 0.f;
    __syncthreads();
    for (int t = 0; t < T_; ++t) {
        float acc = xW[(size_t)xw_stride * t + tid];
        #pragma unroll 8
        for (int k = 0; k < HU_; ++k) acc += hbuf[k] * (float)Wh[k * GH_ + tid];
        gbuf[tid] = acc;
        __syncthreads();
        if (tid < HU_) {
            float ig = gbuf[tid], jg = gbuf[HU_ + tid];
            float fg = gbuf[2 * HU_ + tid], og = gbuf[3 * HU_ + tid];
            c = c * sigm(fg + 1.f) + sigm(ig) * tanhf(jg);
            float h = tanhf(c) * sigm(og);
            hbuf[tid] = h;
            h_out[(size_t)t * HU_ + tid] = h;
        }
        __syncthreads();
    }
}

// ---------------- z0[j] = b[j] + sum_k encoded_h[k] * W[k][j]  (x-part rows) ----
__global__ void __launch_bounds__(800) matvec_z0(const float* hvec, const float* W,
                                                 const float* b, float* z) {
    int j = threadIdx.x;
    float acc = b[j];
    for (int k = 0; k < HU_; ++k) acc += hvec[k] * W[(size_t)k * GH_ + j];
    z[j] = acc;
}

// ---------------- nll_t = log(rowSum[t]) - (outs[t].w[:,y] + sb[y]); sum -> out --
__global__ void __launch_bounds__(256) final_nll(const float* outs, const float* sw,
                                                 const float* sb, const int* sent,
                                                 const float* rowSum, float* out) {
    int t = blockIdx.x * 8 + (threadIdx.x >> 5);
    int lane = threadIdx.x & 31;
    int y = sent[t];
    float acc = 0.f;
    for (int k = lane; k < HU_; k += 32)
        acc += outs[(size_t)t * HU_ + k] * sw[(size_t)k * V_ + y];
    for (int off = 16; off; off >>= 1) acc += __shfl_xor(acc, off, 32);
    if (lane == 0) {
        float nll = logf(rowSum[t]) - (acc + sb[y]);
        atomicAdd(out, nll);
    }
}

// ================================================================================
extern "C" void kernel_launch(void* const* d_in, const int* in_sizes, int n_in,
                              void* d_out, int out_size, void* d_ws, size_t ws_size,
                              hipStream_t stream) {
    const int*   sent  = (const int*)d_in[0];
    const float* emb   = (const float*)d_in[1];
    const float* encW0 = (const float*)d_in[2];
    const float* encB0 = (const float*)d_in[3];
    // d_in[4], d_in[5] (enc layer 1) are dead code in the reference
    const float* decW0 = (const float*)d_in[6];
    const float* decB0 = (const float*)d_in[7];
    const float* decW1 = (const float*)d_in[8];
    const float* decB1 = (const float*)d_in[9];
    const float* sw    = (const float*)d_in[10];
    const float* sb    = (const float*)d_in[11];
    float* out = (float*)d_out;

    char* ws = (char*)d_ws;
    size_t off = 0;
    auto carve = [&](size_t bytes) { void* p = ws + off; off = (off + bytes + 255) & ~(size_t)255; return p; };
    _Float16* wT_vocab = (_Float16*)carve((size_t)NPADV_ * KP_ * 2);   // 22.5 MB
    _Float16* wTx_enc0 = (_Float16*)carve((size_t)NPADX_ * KP_ * 2);
    _Float16* wTx_dec1 = (_Float16*)carve((size_t)NPADX_ * KP_ * 2);
    _Float16* aF16     = (_Float16*)carve((size_t)T_ * KP_ * 2);       // A staging (reused)
    float*    xW       = (float*)carve((size_t)T_ * GH_ * 4);          // input projections (reused)
    float*    h_enc0   = (float*)carve((size_t)T_ * HU_ * 4);
    float*    h_dec0   = (float*)carve((size_t)T_ * HU_ * 4);
    float*    h_dec1   = (float*)carve((size_t)T_ * HU_ * 4);
    float*    z0       = (float*)carve(GH_ * 4);
    float*    rowSum   = (float*)carve(T_ * 4);
    (void)ws_size; (void)n_in; (void)in_sizes; (void)out_size;

    const size_t ldsBytes = (size_t)HU_ * GH_ * 2 + (HU_ + GH_) * 4;   // 324 KB (<320 KiB cap)

    zero_init<<<(T_ + 256) / 256, 256, 0, stream>>>(rowSum, out);
    embed_gather<<<(T_ * KP_ + 255) / 256, 256, 0, stream>>>(sent, emb, aF16);
    wT_cvt<<<(NPADX_ * KP_ + 255) / 256, 256, 0, stream>>>(encW0, wTx_enc0, HU_, GH_, NPADX_);
    wT_cvt<<<(NPADX_ * KP_ + 255) / 256, 256, 0, stream>>>(decW1, wTx_dec1, HU_, GH_, NPADX_);
    wT_cvt<<<(NPADV_ * KP_ + 255) / 256, 256, 0, stream>>>(sw, wT_vocab, HU_, V_, NPADV_);

    // encoder layer 0: input projection (WMMA) then LDS-resident scan
    gemm_bias_f16<<<(T_ / 16 / 8) * NTX_, 256, 0, stream>>>(aF16, wTx_enc0, encB0, xW, GH_, NTX_);
    lstm_layer<<<1, 800, ldsBytes, stream>>>(encW0, xW, GH_, h_enc0);

    // decoder layer 0: constant input -> single matvec projection, then scan
    matvec_z0<<<1, 800, 0, stream>>>(h_enc0 + (size_t)(T_ - 1) * HU_, decW0, decB0, z0);
    lstm_layer<<<1, 800, ldsBytes, stream>>>(decW0, z0, 0, h_dec0);

    // decoder layer 1: input projection of h_dec0 stream (WMMA), then scan
    cvt_pad<<<(T_ * KP_ + 255) / 256, 256, 0, stream>>>(h_dec0, aF16, T_);
    gemm_bias_f16<<<(T_ / 16 / 8) * NTX_, 256, 0, stream>>>(aF16, wTx_dec1, decB1, xW, GH_, NTX_);
    lstm_layer<<<1, 800, ldsBytes, stream>>>(decW1, xW, GH_, h_dec1);

    // logits GEMM (41 GFLOP, compute bound -> WMMA) fused with exp-row-sums
    cvt_pad<<<(T_ * KP_ + 255) / 256, 256, 0, stream>>>(h_dec1, aF16, T_);
    logits_lse<<<(T_ / 16 / 8) * NTV_, 256, 0, stream>>>(aF16, wT_vocab, sb, rowSum);

    // target logits (f32) + log-sum-exp -> summed NLL
    final_nll<<<T_ / 8, 256, 0, stream>>>(h_dec1, sw, sb, sent, rowSum, out);
}
